// Encoder_39659728011570
// MI455X (gfx1250) — compile-verified, matrix operand order checked by reading, not run on hardware
//
#include <hip/hip_runtime.h>
#include <cstddef>

// ---------------------------------------------------------------------------
// Types for CDNA5 WMMA (wave32, 16x16x32 bf16 -> f32 accumulate)
// ---------------------------------------------------------------------------
typedef __attribute__((ext_vector_type(16))) __bf16 v16bf;
typedef __attribute__((ext_vector_type(8)))  float  v8f;

#define CONV_WAVES   4                 // waves per block (one pixel tile each)
#define CONV_THREADS (CONV_WAVES * 32)
#define MTILES       4                 // Cout 16-tiles per block (64 channels)

__device__ __forceinline__ __bf16 f2bf(float f) {
  union { float f; unsigned u; } a; a.f = f;
  unsigned r = a.u + 0x7FFFu + ((a.u >> 16) & 1u);   // round-to-nearest-even
  unsigned short h = (unsigned short)(r >> 16);
  __bf16 b;
  __builtin_memcpy(&b, &h, sizeof(b));
  return b;
}

// ---------------------------------------------------------------------------
// LDS-tiled implicit-GEMM convolution via V_WMMA_F32_16X16X32_BF16.
// Block = 4 waves; macro-tile = 64 Cout x 64 pixels; K-step = 32.
//  - wtile: 64x32 weights -> bf16 in LDS, coalesced, converted once/block.
//  - xtile: 32x64 im2col  -> bf16 in LDS; each thread stages one pixel
//    column (tid&63) for k rows tid>>6, +2, ... (consecutive threads hit
//    consecutive input addresses -> coalesced). The (ci,kh,kw) decomposition
//    is 1 division pair per thread per K-step + carry increments by 2
//    (double-carry also exact for 1x1).
//  - Wave inner loop: 16 B ds-reads + 64 A ds-reads + 4 WMMAs sharing the
//    B fragment (4 f32 accumulators). No global access, no divisions.
// Fragment layouts (ISA 7.12.2):
//  A (16-bit, 16x32): m = lane&15, k(j) = 8*(lane>=16) + (j<8 ? j : j+8)
//  B assumed transpose-symmetric (n = lane&15, same k pattern)
//  D (f32 16x16): n = lane&15, m = r + 8*(lane>=16)
// ---------------------------------------------------------------------------
__global__ __launch_bounds__(CONV_THREADS)
void conv_wmma_bf16(const float* __restrict__ x, const float* __restrict__ w,
                    const float* __restrict__ scale, const float* __restrict__ bias,
                    const float* __restrict__ res, float* __restrict__ y,
                    int N, int Cin, int H, int W, int Cout, int Ho, int Wo,
                    int KH, int KW, int stride, int pad, int relu)
{
  __shared__ __bf16 wtile[MTILES * 16 * 32];   // 64 x 32 bf16 = 4 KB
  __shared__ __bf16 xtile[32 * 64];            // 32 x 64 bf16 = 4 KB

  const int lane = threadIdx.x & 31;
  const int wave = threadIdx.x >> 5;
  const int tid  = threadIdx.x;
  const int hi   = lane >> 4;          // half-wave select
  const int mn   = lane & 15;          // A row (m) and B/D column (n)

  const int coutBase = blockIdx.y * (MTILES * 16);
  const int P        = N * Ho * Wo;
  const int pixBlock = blockIdx.x * 64;
  const int K        = Cin * KH * KW;
  const int Ksteps   = (K + 31) >> 5;
  const int KK       = KH * KW;
  const int HW       = Ho * Wo;

  // ---- compute-side pixel (epilogue store target) ----
  const int pix = pixBlock + (wave << 4) + mn;
  const bool pixOK = pix < P;
  int nimg = 0, oh = 0, ow = 0;
  if (pixOK) {
    nimg = pix / HW;
    int r = pix - nimg * HW;
    oh = r / Wo; ow = r - oh * Wo;
  }

  // ---- staging-side pixel (this thread stages xtile column sp) ----
  const int sp      = tid & 63;        // xtile column (block pixel)
  const int skkBase = tid >> 6;        // first k row (0 or 1), step 2
  const int spix = pixBlock + sp;
  const bool spOK = spix < P;
  int sn = 0, soh = 0, sow = 0;
  if (spOK) {
    sn = spix / HW;
    int r = spix - sn * HW;
    soh = r / Wo; sow = r - soh * Wo;
  }
  const int sih0 = soh * stride - pad;
  const int siw0 = sow * stride - pad;
  const float* __restrict__ xs = x + (size_t)sn * Cin * H * W;

  v8f acc[MTILES] = {};

  for (int ks = 0; ks < Ksteps; ++ks) {
    const int k0 = ks << 5;

    __syncthreads();                        // protect tiles from prev readers

    // ---- stage weights (coalesced, bf16) ----
    for (int t = tid; t < MTILES * 16 * 32; t += CONV_THREADS) {
      int m  = t >> 5;
      int kk = k0 + (t & 31);
      int co = coutBase + m;
      wtile[t] = f2bf((co < Cout && kk < K) ? w[(size_t)co * K + kk] : 0.0f);
    }
    // speculative prefetch of next weight tile -> global_prefetch_b8
    {
      int nk = k0 + 32 + (tid & 31);
      if (nk < K && coutBase < Cout)
        __builtin_prefetch(w + (size_t)coutBase * K + nk, 0, 0);
    }

    // ---- stage im2col tile (coalesced across threads, bf16) ----
    {
      const int kb = k0 + skkBase;
      int ci, kh, kw;
      if (KK == 1) { ci = kb; kh = 0; kw = 0; }
      else { ci = kb / KK; int r = kb - ci * KK; kh = r / KW; kw = r - kh * KW; }
#pragma unroll
      for (int i2 = 0; i2 < 16; ++i2) {
        const int kk = skkBase + (i2 << 1);
        const int k  = k0 + kk;
        float v = 0.0f;
        if (spOK && k < K) {
          int ih = sih0 + kh, iw = siw0 + kw;
          if (ih >= 0 && ih < H && iw >= 0 && iw < W)
            v = xs[((size_t)ci * H + ih) * W + iw];
        }
        xtile[(kk << 6) + sp] = f2bf(v);
        // advance k by 2 with double carry (exact for any KH/KW incl. 1x1)
        kw += 2;
        if (kw >= KW) { kw -= KW; if (++kh >= KH) { kh -= KH; ++ci; } }
        if (kw >= KW) { kw -= KW; if (++kh >= KH) { kh -= KH; ++ci; } }
      }
    }
    __syncthreads();

    // ---- B fragment from LDS ----
    v16bf b;
#pragma unroll
    for (int j = 0; j < 16; ++j) {
      const int kin = (hi << 3) + ((j < 8) ? j : (j + 8));
      b[j] = xtile[(kin << 6) + (wave << 4) + mn];
    }

    // ---- 4 WMMAs sharing the B fragment ----
#pragma unroll
    for (int t = 0; t < MTILES; ++t) {
      v16bf a;
#pragma unroll
      for (int j = 0; j < 16; ++j) {
        const int kin = (hi << 3) + ((j < 8) ? j : (j + 8));
        a[j] = wtile[(((t << 4) + mn) << 5) + kin];
      }
      acc[t] = __builtin_amdgcn_wmma_f32_16x16x32_bf16(
          /*neg_a=*/false, a, /*neg_b=*/false, b,
          /*c_mod=*/(short)0, acc[t], /*reuse_a=*/false, /*reuse_b=*/false);
    }
  }

  // Epilogue: affine (BN or bias) + optional residual + optional ReLU.
#pragma unroll
  for (int t = 0; t < MTILES; ++t) {
#pragma unroll
    for (int r = 0; r < 8; ++r) {
      const int m  = (t << 4) + r + (hi << 3);
      const int co = coutBase + m;
      if (co < Cout && pixOK) {
        float v = acc[t][r];
        float s = scale ? scale[co] : 1.0f;
        float bb = bias ? bias[co] : 0.0f;
        v = v * s + bb;
        size_t oi = (((size_t)nimg * Cout + co) * Ho + oh) * Wo + ow;
        if (res)  v += res[oi];
        if (relu) v = v > 0.0f ? v : 0.0f;
        y[oi] = v;
      }
    }
  }
}

// ---------------------------------------------------------------------------
// Elementwise kernels
// ---------------------------------------------------------------------------
__global__ void normalize_k(const float* __restrict__ x, float* __restrict__ y, int total)
{
  int i = blockIdx.x * blockDim.x + threadIdx.x;
  if (i >= total) return;
  const float mean[3] = {0.485f, 0.456f, 0.406f};
  const float stdv[3] = {0.229f, 0.224f, 0.225f};
  int c = (i >> 16) % 3;                    // HW = 256*256 = 65536
  y[i] = (x[i] - mean[c]) / stdv[c];
}

__global__ void maxpool3x3s2_k(const float* __restrict__ x, float* __restrict__ y,
                               int NC, int H, int W, int Ho, int Wo)
{
  int i = blockIdx.x * blockDim.x + threadIdx.x;
  int total = NC * Ho * Wo;
  if (i >= total) return;
  int wo = i % Wo; int t = i / Wo; int ho = t % Ho; int nc = t / Ho;
  int h0 = ho * 2 - 1, w0 = wo * 2 - 1;
  float m = -3.4e38f;
  for (int dh = 0; dh < 3; ++dh)
    for (int dw = 0; dw < 3; ++dw) {
      int ih = h0 + dh, iw = w0 + dw;
      if (ih >= 0 && ih < H && iw >= 0 && iw < W) {
        float v = x[((size_t)nc * H + ih) * W + iw];
        m = v > m ? v : m;
      }
    }
  y[i] = m;
}

__global__ void avgpool8_k(const float* __restrict__ x, float* __restrict__ y,
                           int NC, int H, int W)
{
  int i = blockIdx.x * blockDim.x + threadIdx.x;
  int total = NC * 64;
  if (i >= total) return;
  int jj = i & 7; int t = i >> 3; int ii = t & 7; int nc = t >> 3;
  int fh = H >> 3, fw = W >> 3;
  float s = 0.0f;
  for (int a = 0; a < fh; ++a)
    for (int b = 0; b < fw; ++b)
      s += x[((size_t)nc * H + ii * fh + a) * W + jj * fw + b];
  y[i] = s / (float)(fh * fw);
}

__global__ void gather_sum_k(const float* __restrict__ e, const int* __restrict__ pts,
                             float* __restrict__ o, int B, int C)
{
  int i = blockIdx.x * blockDim.x + threadIdx.x;
  int total = B * C * 64;
  if (i >= total) return;
  int j = i & 63; int t = i >> 6; int c = t % C; int b = t / C;
  const int* p = pts + ((size_t)(b * 64 + j)) * 10;   // 5 points * 2 coords
  float s = 0.0f;
  for (int q = 0; q < 5; ++q) {
    int r0 = p[q * 2 + 0], r1 = p[q * 2 + 1];
    s += e[(((size_t)b * C + c) * 8 + r0) * 8 + r1];
  }
  o[i] = s;
}

// y = base + bilinear_upsample(src, factor f)   (align_corners=False)
__global__ void resize_add_k(const float* __restrict__ base, const float* __restrict__ src,
                             float* __restrict__ y, int NC, int Hs, int Ws, int f)
{
  int Ho = Hs * f, Wo = Ws * f;
  int i = blockIdx.x * blockDim.x + threadIdx.x;
  int total = NC * Ho * Wo;
  if (i >= total) return;
  int wo = i % Wo; int t = i / Wo; int ho = t % Ho; int nc = t / Ho;
  float out;
  const float* s = src + (size_t)nc * Hs * Ws;
  if (f == 1) {
    out = s[ho * Ws + wo];
  } else {
    float sy = (ho + 0.5f) / (float)f - 0.5f;
    float sx = (wo + 0.5f) / (float)f - 0.5f;
    int y0 = (int)floorf(sy), x0 = (int)floorf(sx);
    float wy = sy - (float)y0, wx = sx - (float)x0;
    int y1 = y0 + 1, x1 = x0 + 1;
    y0 = y0 < 0 ? 0 : (y0 > Hs - 1 ? Hs - 1 : y0);
    y1 = y1 < 0 ? 0 : (y1 > Hs - 1 ? Hs - 1 : y1);
    x0 = x0 < 0 ? 0 : (x0 > Ws - 1 ? Ws - 1 : x0);
    x1 = x1 < 0 ? 0 : (x1 > Ws - 1 ? Ws - 1 : x1);
    out = (1.0f - wy) * ((1.0f - wx) * s[y0 * Ws + x0] + wx * s[y0 * Ws + x1])
        +         wy  * ((1.0f - wx) * s[y1 * Ws + x0] + wx * s[y1 * Ws + x1]);
  }
  y[i] = base[i] + out;
}

// ---------------------------------------------------------------------------
// Host-side orchestration
// ---------------------------------------------------------------------------
namespace {

struct BlkP { const float *w1,*s1,*b1,*w2,*s2,*b2,*wd,*sd,*bd; int down; };
struct NetP { const float *w0,*s0,*b0; BlkP blk[16]; int nb[4]; };
struct FuseP {
  const float *ic_w,*ic_b,*lc_w,*lc_b,*id_w,*id_b,*ld_w,*ld_b;
  const float *ipW[3],*ipB[3],*lpW[3],*lpB[3];
};

const int CH[4] = {64, 128, 256, 512};
const int SCALES[4] = {8, 4, 2, 1};

int walk_net(void* const* din, int idx, NetP& n, const int nb[4]) {
  n.w0 = (const float*)din[idx++];
  n.s0 = (const float*)din[idx++];
  n.b0 = (const float*)din[idx++];
  int bi = 0;
  for (int li = 0; li < 4; ++li) {
    n.nb[li] = nb[li];
    for (int b = 0; b < nb[li]; ++b) {
      BlkP& p = n.blk[bi++];
      p.w1 = (const float*)din[idx++];
      p.s1 = (const float*)din[idx++];
      p.b1 = (const float*)din[idx++];
      p.w2 = (const float*)din[idx++];
      p.s2 = (const float*)din[idx++];
      p.b2 = (const float*)din[idx++];
      p.down = (li > 0 && b == 0) ? 1 : 0;
      if (p.down) {
        p.wd = (const float*)din[idx++];
        p.sd = (const float*)din[idx++];
        p.bd = (const float*)din[idx++];
      } else { p.wd = p.sd = p.bd = nullptr; }
    }
  }
  return idx;
}

int walk_fuse(void* const* din, int idx, FuseP& f) {
  f.ic_w = (const float*)din[idx++]; f.ic_b = (const float*)din[idx++];
  f.lc_w = (const float*)din[idx++]; f.lc_b = (const float*)din[idx++];
  f.id_w = (const float*)din[idx++]; f.id_b = (const float*)din[idx++];
  f.ld_w = (const float*)din[idx++]; f.ld_b = (const float*)din[idx++];
  for (int j = 0; j < 3; ++j) { f.ipW[j] = (const float*)din[idx++]; f.ipB[j] = (const float*)din[idx++]; }
  for (int j = 0; j < 3; ++j) { f.lpW[j] = (const float*)din[idx++]; f.lpB[j] = (const float*)din[idx++]; }
  return idx;
}

void conv(hipStream_t st, const float* x, const float* w,
          const float* s, const float* b, const float* res, float* y,
          int N, int Cin, int H, int W, int Cout,
          int KH, int KW, int stride, int pad, int relu) {
  int Ho = (H + 2 * pad - KH) / stride + 1;
  int Wo = (W + 2 * pad - KW) / stride + 1;
  int P = N * Ho * Wo;
  dim3 grid((P + 63) / 64, (Cout + MTILES * 16 - 1) / (MTILES * 16));
  conv_wmma_bf16<<<grid, CONV_THREADS, 0, st>>>(
      x, w, s, b, res, y, N, Cin, H, W, Cout, Ho, Wo, KH, KW, stride, pad, relu);
}

void run_layer(hipStream_t st, NetP& net, int& bi,
               float*& cur, float*& t1, float*& t2,
               int li, int& H, int Cin) {
  const int C = CH[li];
  for (int b = 0; b < net.nb[li]; ++b) {
    BlkP& p = net.blk[bi++];
    const int stride = (li > 0 && b == 0) ? 2 : 1;
    const int ic = (b == 0) ? Cin : C;
    const int Hout = H / stride;
    // conv1 + bn1 + relu
    conv(st, cur, p.w1, p.s1, p.b1, nullptr, t1, 8, ic, H, H, C, 3, 3, stride, 1, 1);
    if (p.down) {
      // shortcut = bn(down(x))
      conv(st, cur, p.wd, p.sd, p.bd, nullptr, t2, 8, ic, H, H, C, 1, 1, stride, 0, 0);
      // out = relu(bn2(conv2(t1)) + shortcut) -> reuse cur
      conv(st, t1, p.w2, p.s2, p.b2, t2, cur, 8, C, Hout, Hout, C, 3, 3, 1, 1, 1);
      // cur stays cur
    } else {
      conv(st, t1, p.w2, p.s2, p.b2, cur, t2, 8, C, Hout, Hout, C, 3, 3, 1, 1, 1);
      float* tmp = cur; cur = t2; t2 = tmp;
    }
    H = Hout;
  }
}

} // namespace

extern "C" void kernel_launch(void* const* d_in, const int* in_sizes, int n_in,
                              void* d_out, int out_size, void* d_ws, size_t ws_size,
                              hipStream_t stream) {
  (void)in_sizes; (void)n_in; (void)out_size; (void)ws_size;

  const float* image   = (const float*)d_in[0];   // [8,3,256,256]
  const float* lidar   = (const float*)d_in[1];   // [8,2,256,256]
  /* velocity d_in[2] unused by the reference forward pass */
  const int*   bev_pts = (const int*)d_in[3];     // [8,8,8,5,2]
  const int*   img_pts = (const int*)d_in[4];     // [8,8,8,5,2]
  float*       out     = (float*)d_out;           // img_f (262144) ++ lid_f (262144)

  // ---- parameter tables (dict insertion order) ----
  const int IMG_NB[4] = {3, 4, 6, 3};   // ResNet34
  const int LID_NB[4] = {2, 2, 2, 2};   // ResNet18
  NetP  img, lid;
  FuseP fuse[4];
  int idx = 5;
  idx = walk_net(d_in, idx, img, IMG_NB);
  idx = walk_net(d_in, idx, lid, LID_NB);
  for (int i = 0; i < 4; ++i) idx = walk_fuse(d_in, idx, fuse[i]);

  // ---- workspace bump allocator ----
  float* ws = (float*)d_ws;
  size_t off = 0;
  auto A = [&](size_t n) { float* p = ws + off; off += n; return p; };
  float* NORM = A((size_t)8 * 3 * 256 * 256);        // normalized image
  float* BIG  = A((size_t)8 * 64 * 128 * 128);       // stem conv scratch
  float* iA = A(2097152); float* iB = A(2097152); float* iC = A(2097152);
  float* lA = A(2097152); float* lB = A(2097152); float* lC = A(2097152);
  float* sPool = A(262144); float* sEmbI = A(262144); float* sEmbL = A(262144);
  float* sG = A(262144);  float* sM0 = A(262144);  float* sM1 = A(262144);
  float* sP = A(262144);

  const int T = 256;
  auto blocks = [](int n) { return (n + 255) / 256; };

  // ---- stems: conv7x7/s2 + bn + relu + maxpool3x3/s2 ----
  normalize_k<<<blocks(8 * 3 * 65536), T, 0, stream>>>(image, NORM, 8 * 3 * 65536);
  conv(stream, NORM, img.w0, img.s0, img.b0, nullptr, BIG, 8, 3, 256, 256, 64, 7, 7, 2, 3, 1);
  maxpool3x3s2_k<<<blocks(8 * 64 * 64 * 64), T, 0, stream>>>(BIG, iA, 8 * 64, 128, 128, 64, 64);
  conv(stream, lidar, lid.w0, lid.s0, lid.b0, nullptr, BIG, 8, 2, 256, 256, 64, 7, 7, 2, 3, 1);
  maxpool3x3s2_k<<<blocks(8 * 64 * 64 * 64), T, 0, stream>>>(BIG, lA, 8 * 64, 128, 128, 64, 64);

  // ---- 4 layers + fusion ----
  float *icur = iA, *it1 = iB, *it2 = iC;
  float *lcur = lA, *lt1 = lB, *lt2 = lC;
  int iH = 64, lH = 64, ibi = 0, lbi = 0;

  for (int i = 0; i < 4; ++i) {
    const int Cin = (i == 0) ? 64 : CH[i - 1];
    const int Ci = CH[i];
    run_layer(stream, img, ibi, icur, it1, it2, i, iH, Cin);
    run_layer(stream, lid, lbi, lcur, lt1, lt2, i, lH, Cin);
    const int Hc = iH;  // == lH
    const FuseP& f = fuse[i];

    // embeddings: avgpool to 8x8 FIRST (commutes exactly with the 1x1 conv)
    avgpool8_k<<<blocks(8 * Ci * 64), T, 0, stream>>>(icur, sPool, 8 * Ci, Hc, Hc);
    conv(stream, sPool, f.ic_w, nullptr, f.ic_b, nullptr, sEmbI, 8, Ci, 8, 8, 512, 1, 1, 1, 0, 0);
    avgpool8_k<<<blocks(8 * Ci * 64), T, 0, stream>>>(lcur, sPool, 8 * Ci, Hc, Hc);
    conv(stream, sPool, f.lc_w, nullptr, f.lc_b, nullptr, sEmbL, 8, Ci, 8, 8, 512, 1, 1, 1, 0, 0);

    // image -> BEV branch: gather + 3x(512x512 MLP as 1x1 conv, WMMA) + project
    gather_sum_k<<<blocks(8 * 512 * 64), T, 0, stream>>>(sEmbI, bev_pts, sG, 8, 512);
    conv(stream, sG,  f.ipW[0], nullptr, f.ipB[0], nullptr, sM0, 8, 512, 8, 8, 512, 1, 1, 1, 0, 1);
    conv(stream, sM0, f.ipW[1], nullptr, f.ipB[1], nullptr, sM1, 8, 512, 8, 8, 512, 1, 1, 1, 0, 1);
    conv(stream, sM1, f.ipW[2], nullptr, f.ipB[2], nullptr, sM0, 8, 512, 8, 8, 512, 1, 1, 1, 0, 1);
    conv(stream, sM0, f.ld_w, nullptr, f.ld_b, nullptr, sP, 8, 512, 8, 8, Ci, 1, 1, 1, 0, 0);
    float* lout = (i == 3) ? (out + 262144) : lt1;
    resize_add_k<<<blocks(8 * Ci * Hc * Hc), T, 0, stream>>>(lcur, sP, lout, 8 * Ci, 8, 8, SCALES[i]);

    // lidar -> image branch (pre-update lid embedding, as in the source)
    gather_sum_k<<<blocks(8 * 512 * 64), T, 0, stream>>>(sEmbL, img_pts, sG, 8, 512);
    conv(stream, sG,  f.lpW[0], nullptr, f.lpB[0], nullptr, sM0, 8, 512, 8, 8, 512, 1, 1, 1, 0, 1);
    conv(stream, sM0, f.lpW[1], nullptr, f.lpB[1], nullptr, sM1, 8, 512, 8, 8, 512, 1, 1, 1, 0, 1);
    conv(stream, sM1, f.lpW[2], nullptr, f.lpB[2], nullptr, sM0, 8, 512, 8, 8, 512, 1, 1, 1, 0, 1);
    conv(stream, sM0, f.id_w, nullptr, f.id_b, nullptr, sP, 8, 512, 8, 8, Ci, 1, 1, 1, 0, 0);
    float* iout = (i == 3) ? out : it1;
    resize_add_k<<<blocks(8 * Ci * Hc * Hc), T, 0, stream>>>(icur, sP, iout, 8 * Ci, 8, 8, SCALES[i]);

    if (i < 3) {
      float* t;
      t = icur; icur = iout; it1 = t;
      t = lcur; lcur = lout; lt1 = t;
    }
  }
}
